// Word_Att_6519760355537
// MI455X (gfx1250) — matrix operand order repaired and checked by reading, not compiled, not run on hardware
//
#include <hip/hip_runtime.h>
#include <hip/hip_bf16.h>

// ---------------- types ----------------
typedef __bf16 bf16_t;
typedef __attribute__((ext_vector_type(16))) __bf16 v16bf;
typedef __attribute__((ext_vector_type(8)))  float  v8f;
typedef __attribute__((ext_vector_type(4)))  unsigned int v4u;

union FragBF {
    v16bf v;
    v4u   q[2];
};

// Problem constants
#define B_  64
#define H_  1024
#define T_  120
#define TP  128     // padded T
#define Q_  16
#define C_  95
#define CP  96      // padded C
#define DA_ 1024

// ---------------- fragment loaders (bf16, K-contiguous row-major source) ----------------
// A fragment: 16x32 (MxK). lane m = lane&15; half = lane>>4.
//   VGPR0..3: K = half*8 .. half*8+7 ; VGPR4..7: K = 16+half*8 .. +7
__device__ __forceinline__ v16bf load_frag_a(const bf16_t* __restrict__ base, int ld, int lane) {
    const int m    = lane & 15;
    const int half = lane >> 4;
    const bf16_t* row = base + (size_t)m * ld + half * 8;
    FragBF f;
    f.q[0] = *(const v4u*)(row);        // K in [half*8, half*8+8)
    f.q[1] = *(const v4u*)(row + 16);   // K in [16+half*8, 16+half*8+8)
    return f.v;
}

// B fragment: 32x16 (KxN). lane n = lane&15; K = half*8..+7 and 16+half*8..+7 (same as A).
// Source: row n of W (W stored [N, K] row-major, K contiguous).
__device__ __forceinline__ v16bf load_frag_b(const bf16_t* __restrict__ base, int ld, int lane) {
    const int n    = lane & 15;
    const int half = lane >> 4;
    const bf16_t* p = base + (size_t)n * ld + half * 8;
    FragBF f;
    f.q[0] = *(const v4u*)(p);
    f.q[1] = *(const v4u*)(p + 16);
    return f.v;
}

#define WMMA_BF16(a, b, c) \
    __builtin_amdgcn_wmma_f32_16x16x32_bf16(false, (a), false, (b), (short)0, (c), false, false)

// ---------------- pack kernels ----------------
// x [B,H,T,1] fp32 -> xb [B, TP, H] bf16 (transpose per batch, zero-pad t in [120,128))
__global__ __launch_bounds__(256) void pack_x_kernel(const float* __restrict__ x,
                                                     bf16_t* __restrict__ xb) {
    __shared__ float tile[32][33];
    const int t0 = blockIdx.x * 32;   // 0,32,64,96
    const int h0 = blockIdx.y * 32;   // 0..992
    const int b  = blockIdx.z;
    const int tx = threadIdx.x & 31;
    const int ty = threadIdx.x >> 5;  // 0..7
    #pragma unroll
    for (int i = 0; i < 32; i += 8) {
        const int h = h0 + ty + i;
        const int t = t0 + tx;
        float v = 0.0f;
        if (t < T_) v = x[((size_t)b * H_ + h) * T_ + t];   // coalesced over t
        tile[ty + i][tx] = v;
    }
    __syncthreads();
    #pragma unroll
    for (int i = 0; i < 32; i += 8) {
        const int t = t0 + ty + i;
        const int h = h0 + tx;
        xb[((size_t)b * TP + t) * H_ + h] = (bf16_t)tile[tx][ty + i];  // coalesced over h
    }
}

// w1 [DA, H] fp32 -> bf16 (same layout)
__global__ __launch_bounds__(256) void pack_w1_kernel(const float* __restrict__ w1,
                                                      bf16_t* __restrict__ w1b) {
    const int idx = blockIdx.x * 256 + threadIdx.x;   // DA_*H_ = 1M elements
    w1b[idx] = (bf16_t)w1[idx];
}

// w2 [C, DA] fp32 -> [CP, DA] bf16, row 95 zeroed
__global__ __launch_bounds__(256) void pack_w2_kernel(const float* __restrict__ w2,
                                                      bf16_t* __restrict__ w2b) {
    const int idx = blockIdx.x * 256 + threadIdx.x;   // CP*DA_ = 98304
    const int c = idx / DA_;
    w2b[idx] = (c < C_) ? (bf16_t)w2[idx] : (bf16_t)0.0f;
}

// ---------------- GEMM1: a1[b,t,d] = sum_h xb[b,t,h] * w1[d,h]  (bf16 in, bf16 out) ----------------
// Wave = 1 M-tile (16 rows of t) x 4 N-tiles (64 d). Waves: B * 8 * 16 = 8192 -> 1024 blocks.
// K-loop is software-pipelined: iteration k issues loads for k+32 before running WMMAs on k's
// fragments, so VMEM stays in flight behind the matrix ops instead of s_wait_loadcnt 0 per WMMA.
__global__ __launch_bounds__(256) void gemm1_kernel(const bf16_t* __restrict__ xb,
                                                    const bf16_t* __restrict__ w1b,
                                                    bf16_t* __restrict__ a1b) {
    const int lane = threadIdx.x & 31;
    const int wave = threadIdx.x >> 5;
    const int w  = blockIdx.x * 8 + wave;
    const int b  = w >> 7;          // / (8*16)
    const int r  = w & 127;
    const int mt = r >> 4;          // m-tile 0..7
    const int ng = r & 15;          // n-group 0..15 (64 d's each)

    const bf16_t* A  = xb  + ((size_t)b * TP + mt * 16) * H_;
    const bf16_t* Bw = w1b + (size_t)(ng * 64) * H_;

    v8f acc0 = {}, acc1 = {}, acc2 = {}, acc3 = {};

    // prologue loads (k = 0)
    v16bf af = load_frag_a(A, H_, lane);
    v16bf b0 = load_frag_b(Bw,            H_, lane);
    v16bf b1 = load_frag_b(Bw + 16 * H_,  H_, lane);
    v16bf b2 = load_frag_b(Bw + 32 * H_,  H_, lane);
    v16bf b3 = load_frag_b(Bw + 48 * H_,  H_, lane);

    #pragma unroll 2
    for (int k = 32; k < H_; k += 32) {
        // issue next iteration's loads
        const v16bf afn = load_frag_a(A + k, H_, lane);
        const v16bf c0  = load_frag_b(Bw + k,            H_, lane);
        const v16bf c1  = load_frag_b(Bw + 16 * H_ + k,  H_, lane);
        const v16bf c2  = load_frag_b(Bw + 32 * H_ + k,  H_, lane);
        const v16bf c3  = load_frag_b(Bw + 48 * H_ + k,  H_, lane);
        // compute on previous iteration's fragments (independent of the loads above)
        acc0 = WMMA_BF16(af, b0, acc0);
        acc1 = WMMA_BF16(af, b1, acc1);
        acc2 = WMMA_BF16(af, b2, acc2);
        acc3 = WMMA_BF16(af, b3, acc3);
        af = afn; b0 = c0; b1 = c1; b2 = c2; b3 = c3;
    }
    // epilogue
    acc0 = WMMA_BF16(af, b0, acc0);
    acc1 = WMMA_BF16(af, b1, acc1);
    acc2 = WMMA_BF16(af, b2, acc2);
    acc3 = WMMA_BF16(af, b3, acc3);

    // D layout: element (m = j + 8*(lane>>4), n = lane&15) in VGPR j
    bf16_t* out = a1b + ((size_t)b * TP + mt * 16) * DA_;
    const int n0 = ng * 64 + (lane & 15);
    const int mh = (lane >> 4) * 8;
    #pragma unroll
    for (int j = 0; j < 8; ++j) {
        const size_t row = (size_t)(j + mh) * DA_;
        out[row + n0 +  0] = (bf16_t)acc0[j];
        out[row + n0 + 16] = (bf16_t)acc1[j];
        out[row + n0 + 32] = (bf16_t)acc2[j];
        out[row + n0 + 48] = (bf16_t)acc3[j];
    }
}

// ---------------- GEMM2: a2t[b,c,t] = sum_d a1[b,t,d] * w2[c,d]  (fp32 out, T-major) ----------------
// Wave = 1 M-tile x 6 N-tiles (all of CP=96). Waves: B * 8 = 512 -> 64 blocks. Same pipelining.
__global__ __launch_bounds__(256) void gemm2_kernel(const bf16_t* __restrict__ a1b,
                                                    const bf16_t* __restrict__ w2b,
                                                    float* __restrict__ a2t) {
    const int lane = threadIdx.x & 31;
    const int wave = threadIdx.x >> 5;
    const int w  = blockIdx.x * 8 + wave;
    const int b  = w >> 3;
    const int mt = w & 7;

    const bf16_t* A = a1b + ((size_t)b * TP + mt * 16) * DA_;

    v8f acc[6];
    #pragma unroll
    for (int n = 0; n < 6; ++n) acc[n] = (v8f){};

    v16bf af = load_frag_a(A, DA_, lane);
    v16bf bf[6];
    #pragma unroll
    for (int n = 0; n < 6; ++n) bf[n] = load_frag_b(w2b + (size_t)(n * 16) * DA_, DA_, lane);

    #pragma unroll 2
    for (int k = 32; k < DA_; k += 32) {
        const v16bf afn = load_frag_a(A + k, DA_, lane);
        v16bf bn[6];
        #pragma unroll
        for (int n = 0; n < 6; ++n)
            bn[n] = load_frag_b(w2b + (size_t)(n * 16) * DA_ + k, DA_, lane);
        #pragma unroll
        for (int n = 0; n < 6; ++n) acc[n] = WMMA_BF16(af, bf[n], acc[n]);
        af = afn;
        #pragma unroll
        for (int n = 0; n < 6; ++n) bf[n] = bn[n];
    }
    #pragma unroll
    for (int n = 0; n < 6; ++n) acc[n] = WMMA_BF16(af, bf[n], acc[n]);

    const int nlo = lane & 15;
    const int mh  = (lane >> 4) * 8;
    #pragma unroll
    for (int n = 0; n < 6; ++n) {
        const int c = n * 16 + nlo;
        float* p = a2t + ((size_t)b * CP + c) * TP + mt * 16 + mh;
        #pragma unroll
        for (int j = 0; j < 8; ++j) p[j] = acc[n][j];
    }
}

// ---------------- softmax over T per (b,c) row; one wave32 per row, in place ----------------
__global__ __launch_bounds__(256) void softmax_kernel(float* __restrict__ a2t) {
    const int row = blockIdx.x * 8 + (threadIdx.x >> 5);
    if (row >= B_ * C_) return;
    const int b = row / C_;
    const int c = row % C_;
    float* p = a2t + ((size_t)b * CP + c) * TP;
    const int lane = threadIdx.x & 31;

    float v[4];
    float mx = -3.402823466e+38f;
    #pragma unroll
    for (int i = 0; i < 4; ++i) {
        const int t = lane + i * 32;
        v[i] = (t < T_) ? p[t] : -3.402823466e+38f;
        mx = fmaxf(mx, v[i]);
    }
    #pragma unroll
    for (int off = 16; off > 0; off >>= 1) mx = fmaxf(mx, __shfl_xor(mx, off, 32));

    float s = 0.0f;
    #pragma unroll
    for (int i = 0; i < 4; ++i) {
        const int t = lane + i * 32;
        const float e = (t < T_) ? __expf(v[i] - mx) : 0.0f;
        v[i] = e;
        s += e;
    }
    #pragma unroll
    for (int off = 16; off > 0; off >>= 1) s += __shfl_xor(s, off, 32);
    const float inv = 1.0f / s;
    #pragma unroll
    for (int i = 0; i < 4; ++i) {
        const int t = lane + i * 32;
        if (t < T_) p[t] = v[i] * inv;
    }
}

// ---------------- output: out[b,q,h,t] = x[b,h,t] * a4[b, aq[b,q], t] ----------------
// 503 MB stream; t-contiguous coalesced. 2 rows of 120 per 256-thread block.
__global__ __launch_bounds__(256) void out_kernel(const float* __restrict__ x,
                                                  const float* __restrict__ a2t,
                                                  const int* __restrict__ aq,
                                                  float* __restrict__ out) {
    const long long rid = (long long)blockIdx.x * 2 + (threadIdx.x >> 7);
    const int t = threadIdx.x & 127;
    if (t >= T_) return;
    const int h = (int)(rid & (H_ - 1));
    const long long r2 = rid >> 10;
    const int q = (int)(r2 & (Q_ - 1));
    const int b = (int)(r2 >> 4);
    const int c = aq[b * Q_ + q];
    const float g  = a2t[((size_t)b * CP + c) * TP + t];
    const float xv = x[((size_t)b * H_ + h) * T_ + t];
    out[(size_t)rid * T_ + t] = xv * g;
}

// ---------------- launch ----------------
extern "C" void kernel_launch(void* const* d_in, const int* in_sizes, int n_in,
                              void* d_out, int out_size, void* d_ws, size_t ws_size,
                              hipStream_t stream) {
    const float* x  = (const float*)d_in[0];   // [B,H,T,1]
    const float* w1 = (const float*)d_in[1];   // [DA,H]
    const float* w2 = (const float*)d_in[2];   // [C,DA]
    const int*   aq = (const int*)d_in[3];     // [B,Q]
    // d_in[4] mask, d_in[5] bag_ids: unused by reference math
    float* out = (float*)d_out;

    char* ws = (char*)d_ws;
    bf16_t* xb  = (bf16_t*)(ws);                    // B*TP*H  bf16 = 16 MB
    bf16_t* a1b = (bf16_t*)(ws + (16u << 20));      // B*TP*DA bf16 = 16 MB
    bf16_t* w1b = (bf16_t*)(ws + (32u << 20));      // DA*H    bf16 =  2 MB
    bf16_t* w2b = (bf16_t*)(ws + (34u << 20));      // CP*DA   bf16 = 192 KB
    float*  a2t = (float*) (ws + (35u << 20));      // B*CP*TP f32  =  3 MB

    // pack
    pack_x_kernel <<<dim3(4, 32, B_), 256, 0, stream>>>(x, xb);
    pack_w1_kernel<<<(DA_ * H_) / 256, 256, 0, stream>>>(w1, w1b);
    pack_w2_kernel<<<(CP * DA_) / 256, 256, 0, stream>>>(w2, w2b);

    // GEMM1: B*8*16 waves / 8 waves per block = 1024 blocks
    gemm1_kernel<<<1024, 256, 0, stream>>>(xb, w1b, a1b);
    // GEMM2: B*8 waves / 8 = 64 blocks
    gemm2_kernel<<<64, 256, 0, stream>>>(a1b, w2b, a2t);

    // softmax: B*C rows, 8 per block
    softmax_kernel<<<(B_ * C_ + 7) / 8, 256, 0, stream>>>(a2t);

    // output: B*Q*H rows, 2 per block
    out_kernel<<<(B_ * Q_ * H_) / 2, 256, 0, stream>>>(x, a2t, aq, out);
}
